// GELU70_17566416240682
// MI455X (gfx1250) — matrix operand order repaired and checked by reading, not compiled, not run on hardware
//
#include <hip/hip_runtime.h>
#include <hip/hip_bf16.h>

#define BATCH 8
#define TLEN  2048
#define DDIM  2048
#define NSTILES 26   // band width = 16*(NSTILES-1) = 400; d^400 underflows fp32

typedef __bf16 bf16;
typedef bf16  v16bf __attribute__((ext_vector_type(16)));
typedef float v8f   __attribute__((ext_vector_type(8)));

union V16 { v16bf v; uint4 u[2]; };

__device__ __forceinline__ float gelu_f(float v) {
    const float c = 0.7978845608028654f;   // sqrt(2/pi)
    float t = tanhf(c * (v + 0.044715f * v * v * v));
    return 0.5f * v * (1.0f + t);
}

// ---------------------------------------------------------------- k_init
// scal[0] = ||ema_slow||, scal[1] = 0 (gate-sum accumulator)
__global__ void k_init(const float* __restrict__ es, float* __restrict__ scal) {
    int tid = threadIdx.x;
    float ss = 0.f;
    #pragma unroll
    for (int e = 0; e < DDIM / 256; ++e) {
        float v = es[tid + e * 256];
        ss += v * v;
    }
    for (int m = 16; m; m >>= 1) ss += __shfl_xor(ss, m, 32);
    __shared__ float sred[8];
    int wave = tid >> 5, lane = tid & 31;
    if (lane == 0) sred[wave] = ss;
    __syncthreads();
    if (tid == 0) {
        float S = 0.f;
        for (int w = 0; w < 8; ++w) S += sred[w];
        scal[0] = sqrtf(S);
        scal[1] = 0.f;
    }
}

// ---------------------------------------------------------------- k_gelu_stats
// Per (b,t) row: out=gelu(x), rr=||out||, p0=<out, ema_slow>, outb=bf16(out)
__global__ void k_gelu_stats(const float* __restrict__ x, const float* __restrict__ es,
                             bf16* __restrict__ outb, float* __restrict__ rr,
                             float* __restrict__ p0) {
    int row = blockIdx.x;
    int tid = threadIdx.x;
    const float* xr = x + (size_t)row * DDIM;
    bf16* ob = outb + (size_t)row * DDIM;
    float ss = 0.f, dp = 0.f;
    #pragma unroll
    for (int e = 0; e < DDIM / 256; ++e) {
        int i = tid + e * 256;
        float g = gelu_f(xr[i]);
        ss += g * g;
        dp += g * es[i];
        ob[i] = (bf16)g;
    }
    for (int m = 16; m; m >>= 1) {
        ss += __shfl_xor(ss, m, 32);
        dp += __shfl_xor(dp, m, 32);
    }
    __shared__ float sred[16];
    int wave = tid >> 5, lane = tid & 31;
    if (lane == 0) { sred[wave] = ss; sred[8 + wave] = dp; }
    __syncthreads();
    if (tid == 0) {
        float S = 0.f, P = 0.f;
        for (int w = 0; w < 8; ++w) { S += sred[w]; P += sred[8 + w]; }
        rr[row] = sqrtf(S);
        p0[row] = P;
    }
}

// ---------------------------------------------------------------- k_band_gram
// One wave per 16-row tile. M[b,t] = sum_{s<t, t-s<=band} d^(t-1-s) * <out[t], out[s]>
// Banded Gram via v_wmma_f32_16x16x32_bf16; two s-tiles per K-pass share the A loads.
__global__ void k_band_gram(const bf16* __restrict__ outb, const float* __restrict__ p_ldf,
                            float* __restrict__ Mout) {
    const int lane  = threadIdx.x & 31;
    const int wavei = blockIdx.x * (blockDim.x >> 5) + (threadIdx.x >> 5);
    const int b     = wavei >> 7;            // 128 row-tiles per batch
    const int t0    = (wavei & 127) << 4;
    const int half  = lane >> 4;
    const int r     = lane & 15;             // A-row-in-tile == C column index j

    float ldf = p_ldf[0];
    float d   = 1.f / (1.f + __expf(-ldf)); // sigmoid
    float l2d = __log2f(fmaxf(d, 1e-30f));

    // A operand: 16-bit A 16x32 layout — lane holds row t0+r, K chunks [8h..8h+7], [8h+16..8h+23]
    const bf16* Abase = outb + ((size_t)(b * TLEN + t0 + r)) * DDIM + half * 8;

    float macc[8];
    #pragma unroll
    for (int v = 0; v < 8; ++v) macc[v] = 0.f;

    for (int jj = 0; jj < NSTILES; jj += 2) {
        int s0a = t0 - (jj << 4);
        if (s0a <= -16) break;               // wave-uniform: both tiles below row 0
        int s0b = s0a - 16;
        int srow_a = s0a + r;
        int srow_b = s0b + r;
        bool sva = (srow_a >= 0);
        bool svb = (srow_b >= 0);
        // B operand: lane holds column s0+r, K chunks [8h..8h+7], [8h+16..8h+23]
        const bf16* Bba = outb + ((size_t)(b * TLEN + (sva ? srow_a : 0))) * DDIM + half * 8;
        const bf16* Bbb = outb + ((size_t)(b * TLEN + (svb ? srow_b : 0))) * DDIM + half * 8;

        v8f ca = {0.f, 0.f, 0.f, 0.f, 0.f, 0.f, 0.f, 0.f};
        v8f cb = {0.f, 0.f, 0.f, 0.f, 0.f, 0.f, 0.f, 0.f};
        #pragma unroll 4
        for (int k = 0; k < DDIM; k += 32) {
            V16 a, b1, b2;
            a.u[0]  = *(const uint4*)(Abase + k);
            a.u[1]  = *(const uint4*)(Abase + k + 16);
            b1.u[0] = *(const uint4*)(Bba + k);
            b1.u[1] = *(const uint4*)(Bba + k + 16);
            b2.u[0] = *(const uint4*)(Bbb + k);
            b2.u[1] = *(const uint4*)(Bbb + k + 16);
            ca = __builtin_amdgcn_wmma_f32_16x16x32_bf16(
                     false, a.v, false, b1.v, (short)0, ca, false, false);
            cb = __builtin_amdgcn_wmma_f32_16x16x32_bf16(
                     false, a.v, false, b2.v, (short)0, cb, false, false);
        }

        // C layout: element (i, j) with i = v + 8*half, j = lane&15
        int bga = t0 - s0a - 1;              // gap = (t-1-s) = bg + i - j
        #pragma unroll
        for (int v = 0; v < 8; ++v) {
            int i = v + (half << 3);
            int ga = bga + i - r;
            int gb = ga + 16;
            // branch-free: raw v_exp_f32, select-to-zero (underflow -> 0 is desired)
            float wa = __builtin_amdgcn_exp2f((float)ga * l2d);
            float wb = __builtin_amdgcn_exp2f((float)gb * l2d);
            wa = (ga >= 0 && sva) ? wa : 0.f;
            wb = (gb >= 0 && svb) ? wb : 0.f;
            macc[v] = fmaf(wa, ca[v], macc[v]);
            macc[v] = fmaf(wb, cb[v], macc[v]);
        }
    }

    // reduce over the 16 columns within each lane-half (masks 1..8 stay in-half)
    for (int mm = 1; mm < 16; mm <<= 1) {
        #pragma unroll
        for (int v = 0; v < 8; ++v) macc[v] += __shfl_xor(macc[v], mm, 32);
    }
    if (r == 0) {
        #pragma unroll
        for (int v = 0; v < 8; ++v)
            Mout[(size_t)b * TLEN + t0 + v + (half << 3)] = macc[v];
    }
}

// ---------------------------------------------------------------- k_scan
// O(1)-per-step scalar recurrence per batch. Emits gate_raw and its sum.
__global__ void k_scan(const float* __restrict__ rr, const float* __restrict__ p0,
                       const float* __restrict__ Mw, float* __restrict__ scal,
                       float* __restrict__ gate_raw,
                       const float* __restrict__ p_ldf, const float* __restrict__ p_lts,
                       const float* __restrict__ p_ltf) {
    int b = threadIdx.x;
    if (b >= BATCH) return;
    float d     = 1.f / (1.f + __expf(-p_ldf[0]));
    float tau_s = __expf(p_lts[0]);
    float tau_f = __expf(p_ltf[0]);
    float omd   = 1.f - d;
    float esn    = scal[0];
    float es_div = fmaxf(esn, 1e-12f);
    float n2     = esn * esn;                // ||ema0||^2
    float dpow   = 1.f;                      // d^t
    float local  = 0.f;
    for (int t = 0; t < TLEN; ++t) {
        int idx  = b * TLEN + t;
        float rt = rr[idx];
        float pp = p0[idx];
        float mm = Mw[idx];
        float rdiv = fmaxf(rt, 1e-12f);
        float craw = dpow * pp + omd * mm;   // <out[t], e[t]>
        float en   = fmaxf(sqrtf(n2), 1e-12f);
        float cf   = fminf(fmaxf(craw / (rdiv * en), -1.f), 1.f);
        float cs   = fminf(fmaxf(pp / (rdiv * es_div), -1.f), 1.f);
        float g    = __expf(-tau_s * cs) * __expf(-tau_f * cf);
        gate_raw[idx] = g;
        local += g;
        n2 = d * d * n2 + 2.f * d * omd * craw + omd * omd * rt * rt;
        dpow *= d;
    }
    atomicAdd(&scal[1], local);
}

// ---------------------------------------------------------------- k_final
// out = gelu(x) * gate_raw[row] / max(mean(gate_raw), 1e-5), float4-vectorized.
__global__ void k_final(const float* __restrict__ x, const float* __restrict__ gate_raw,
                        const float* __restrict__ scal, float* __restrict__ out) {
    int i4 = blockIdx.x * blockDim.x + threadIdx.x;
    if (i4 >= (BATCH * TLEN * DDIM) / 4) return;
    float mean = fmaxf(scal[1] * (1.0f / (BATCH * TLEN)), 1e-5f);
    int row = i4 >> 9;                       // DDIM/4 = 512 float4 per row
    float g = gate_raw[row] / mean;
    float4 xv = ((const float4*)x)[i4];
    float4 o;
    o.x = gelu_f(xv.x) * g;
    o.y = gelu_f(xv.y) * g;
    o.z = gelu_f(xv.z) * g;
    o.w = gelu_f(xv.w) * g;
    ((float4*)out)[i4] = o;
}

// ---------------------------------------------------------------- launcher
extern "C" void kernel_launch(void* const* d_in, const int* in_sizes, int n_in,
                              void* d_out, int out_size, void* d_ws, size_t ws_size,
                              hipStream_t stream) {
    const float* x     = (const float*)d_in[0];
    const float* es    = (const float*)d_in[1];
    // d_in[2] = logit_decay_slow: unused by the reference output
    const float* p_ldf = (const float*)d_in[3];
    const float* p_lts = (const float*)d_in[4];
    const float* p_ltf = (const float*)d_in[5];
    float* outp = (float*)d_out;

    char* ws = (char*)d_ws;
    const size_t OFF_BF = 0;                               // 67,108,864 B bf16(out)
    const size_t OFF_R  = OFF_BF + (size_t)BATCH * TLEN * DDIM * 2;
    const size_t OFF_P0 = OFF_R  + (size_t)BATCH * TLEN * 4;
    const size_t OFF_M  = OFF_P0 + (size_t)BATCH * TLEN * 4;
    const size_t OFF_GR = OFF_M  + (size_t)BATCH * TLEN * 4;
    const size_t OFF_SC = OFF_GR + (size_t)BATCH * TLEN * 4;
    bf16*  outb = (bf16*)(ws + OFF_BF);
    float* rr   = (float*)(ws + OFF_R);
    float* p0   = (float*)(ws + OFF_P0);
    float* Mw   = (float*)(ws + OFF_M);
    float* gr   = (float*)(ws + OFF_GR);
    float* scal = (float*)(ws + OFF_SC);

    k_init<<<1, 256, 0, stream>>>(es, scal);
    k_gelu_stats<<<BATCH * TLEN, 256, 0, stream>>>(x, es, outb, rr, p0);
    // 1024 waves total = 8 batches * 128 row-tiles; 4 waves per block
    k_band_gram<<<(BATCH * (TLEN / 16)) / 4, 128, 0, stream>>>(outb, p_ldf, Mw);
    k_scan<<<1, 32, 0, stream>>>(rr, p0, Mw, scal, gr, p_ldf, p_lts, p_ltf);
    k_final<<<(BATCH * TLEN * DDIM / 4) / 256, 256, 0, stream>>>(x, gr, scal, outp);
}
